// Net_23587960389775
// MI455X (gfx1250) — matrix operand order repaired and checked by reading, not compiled, not run on hardware
//
#include <hip/hip_runtime.h>
#include <hip/hip_bf16.h>
#include <math.h>

// ---------------------------------------------------------------------------
// CDNA5 (gfx1250) SplineCNN U-Net forward.
// All dense contractions run on v_wmma_f32_16x16x32_bf16 (bf16 in, f32 acc).
// Contraction axes are padded to multiples of 32 so every GEMM uses the
// vectorized guard-free fragment loads. Weights pre-transposed to (cout x K).
// ---------------------------------------------------------------------------

typedef __attribute__((ext_vector_type(16))) __bf16 v16bf;
typedef __attribute__((ext_vector_type(8)))  float  v8f;
typedef __bf16 bf16_t;

#define KQ  5
#define K3Q 125

static __device__ __forceinline__ bf16_t f2bf(float f) {
  // round-to-nearest-even f32 -> bf16
  unsigned int u = __float_as_uint(f);
  unsigned int r = (u + 0x7fffu + ((u >> 16) & 1u)) >> 16;
  return __builtin_bit_cast(bf16_t, (unsigned short)r);
}

static __device__ __forceinline__ float eluf(float v) {
  return v > 0.0f ? v : (__expf(v) - 1.0f);
}

// --------------------------- small utility kernels -------------------------

// in: K x C row-major f32  ->  out: C x Kld row-major bf16 (transposed, padded)
__global__ void f32_to_bf16_T_kernel(const float* __restrict__ in, bf16_t* __restrict__ outT,
                                     int K, int C, int Kld) {
  int i = blockIdx.x * blockDim.x + threadIdx.x;
  if (i >= K * C) return;
  int k = i / C, c = i - k * C;
  outT[(long)c * Kld + k] = f2bf(in[i]);
}

__global__ void fill_kernel(float* __restrict__ p, float v, int n) {
  int i = blockIdx.x * blockDim.x + threadIdx.x;
  if (i < n) p[i] = v;
}

// xp (n x 32, pre-zeroed): column 0 = x
__global__ void pad_x_kernel(const float* __restrict__ x, float* __restrict__ xp, int n) {
  int i = blockIdx.x * blockDim.x + threadIdx.x;
  if (i < n) xp[(size_t)i * 32] = x[i];
}

__global__ void pseudo_kernel(const float* __restrict__ pos, const int* __restrict__ row,
                              const int* __restrict__ col, float* __restrict__ ps, int E) {
  int e = blockIdx.x * blockDim.x + threadIdx.x;
  if (e >= E) return;
  int r = row[e], c = col[e];
  float d0 = pos[c * 3 + 0] - pos[r * 3 + 0];
  float d1 = pos[c * 3 + 1] - pos[r * 3 + 1];
  float d2 = pos[c * 3 + 2] - pos[r * 3 + 2];
  float m = fmaxf(fabsf(d0), fmaxf(fabsf(d1), fabsf(d2)));
  float inv = m > 0.0f ? 0.5f / m : 0.0f;
  ps[e * 3 + 0] = d0 * inv + 0.5f;
  ps[e * 3 + 1] = d1 * inv + 0.5f;
  ps[e * 3 + 2] = d2 * inv + 0.5f;
}

__global__ void deg_count_kernel(const int* __restrict__ row, float* __restrict__ deg, int E) {
  int i = blockIdx.x * blockDim.x + threadIdx.x;
  if (i < E) atomicAdd(&deg[row[i]], 1.0f);
}

// ------------------------------ spline scatter -----------------------------
// A[row - rstart, kidx, c] += b * x[col, c];  A row stride = Apad elements.

static __device__ __forceinline__ void spline_weights(const float* __restrict__ ps, int e,
                                                      float fr[3], int kl[3]) {
#pragma unroll
  for (int d = 0; d < 3; ++d) {
    float p = ps[e * 3 + d] * (float)(KQ - 1);
    float kf = floorf(p);
    kf = fminf(fmaxf(kf, 0.0f), (float)(KQ - 2));
    kl[d] = (int)kf;
    fr[d] = p - kf;
  }
}

// block per edge, thread per channel (cin >= 32)
__global__ void spline_scatter_kernel(const float* __restrict__ x, const int* __restrict__ row,
                                      const int* __restrict__ col, const float* __restrict__ ps,
                                      float* __restrict__ A, int rstart, int rcount,
                                      int E, int cin, int Apad) {
  int e = blockIdx.x;
  if (e >= E) return;
  int r = row[e];
  if (r < rstart || r >= rstart + rcount) return;
  int c = threadIdx.x;
  if (c >= cin) return;
  float fr[3]; int kl[3];
  spline_weights(ps, e, fr, kl);
  float xv = x[(long)col[e] * cin + c];
  float* Arow = A + (size_t)(r - rstart) * Apad + c;
#pragma unroll
  for (int s = 0; s < 8; ++s) {
    int b0 = (s >> 2) & 1, b1 = (s >> 1) & 1, b2 = s & 1;
    float w = (b0 ? fr[0] : 1.0f - fr[0]) *
              (b1 ? fr[1] : 1.0f - fr[1]) *
              (b2 ? fr[2] : 1.0f - fr[2]);
    int kidx = ((kl[0] + b0) * KQ + (kl[1] + b1)) * KQ + (kl[2] + b2);
    atomicAdd(Arow + (long)kidx * cin, w * xv);
  }
}

// thread per edge (cin == 1)
__global__ void spline_scatter_c1_kernel(const float* __restrict__ x, const int* __restrict__ row,
                                         const int* __restrict__ col, const float* __restrict__ ps,
                                         float* __restrict__ A, int rstart, int rcount,
                                         int E, int Apad) {
  int e = blockIdx.x * blockDim.x + threadIdx.x;
  if (e >= E) return;
  int r = row[e];
  if (r < rstart || r >= rstart + rcount) return;
  float fr[3]; int kl[3];
  spline_weights(ps, e, fr, kl);
  float xv = x[col[e]];
  float* Arow = A + (size_t)(r - rstart) * Apad;
#pragma unroll
  for (int s = 0; s < 8; ++s) {
    int b0 = (s >> 2) & 1, b1 = (s >> 1) & 1, b2 = s & 1;
    float w = (b0 ? fr[0] : 1.0f - fr[0]) *
              (b1 ? fr[1] : 1.0f - fr[1]) *
              (b2 ? fr[2] : 1.0f - fr[2]);
    int kidx = ((kl[0] + b0) * KQ + (kl[1] + b1)) * KQ + (kl[2] + b2);
    atomicAdd(Arow + kidx, w * xv);
  }
}

// ----------------------------- WMMA GEMM kernel ----------------------------
// Fragment layouts used here (wave32; contiguous k order, not the ISA table order):
//  A (16x32 bf16): lane lo = row; k = j + 16*hi  (contiguous in k)
//  B (32x16 bf16): lane lo = col; k = j + 16*hi  (contiguous in k)
//  C/D (16x16 f32): vgpr i -> row i + 8*hi, col = lo
// Requires Ka % 32 == 0 (and Kr % 32 == 0 when ROOT): buffers are padded.

static __device__ __forceinline__ v16bf load_a_frag_vec(const float* __restrict__ arow,
                                                        int k0, int hi) {
  const float4* p0 = (const float4*)(arow + k0 + (hi << 4));
  const float4* p1 = (const float4*)(arow + k0 + 8 + (hi << 4));
  float4 r0 = p0[0], r1 = p0[1], r2 = p1[0], r3 = p1[1];
  v16bf a;
  a[0]  = f2bf(r0.x); a[1]  = f2bf(r0.y); a[2]  = f2bf(r0.z); a[3]  = f2bf(r0.w);
  a[4]  = f2bf(r1.x); a[5]  = f2bf(r1.y); a[6]  = f2bf(r1.z); a[7]  = f2bf(r1.w);
  a[8]  = f2bf(r2.x); a[9]  = f2bf(r2.y); a[10] = f2bf(r2.z); a[11] = f2bf(r2.w);
  a[12] = f2bf(r3.x); a[13] = f2bf(r3.y); a[14] = f2bf(r3.z); a[15] = f2bf(r3.w);
  return a;
}

template <int NT, bool ROOT>
__global__ __launch_bounds__(32)
void wmma_gemm_vec_kernel(const float* __restrict__ A, const bf16_t* __restrict__ Wt,
                          const float* __restrict__ deg, const float* __restrict__ Xr,
                          const bf16_t* __restrict__ Wrt, const float* __restrict__ bias,
                          float* __restrict__ out, int n, int Ka, int Kr, int do_elu) {
  constexpr int COUT = NT * 16;
  const int lane = threadIdx.x & 31;
  const int tm = blockIdx.x << 4;
  const int hi = lane >> 4;
  const int lo = lane & 15;
  int mrow = tm + lo;
  if (mrow >= n) mrow = n - 1;         // clamp; masked on store
  const float* arow = A + (size_t)mrow * Ka;

  v8f acc[NT];
#pragma unroll
  for (int t = 0; t < NT; ++t) acc[t] = (v8f)(0.0f);

  for (int k0 = 0; k0 < Ka; k0 += 32) {
    if (k0 + 256 < Ka)                       // speculative prefetch of the A stream
      __builtin_prefetch(arow + k0 + 256, 0, 0);
    v16bf a = load_a_frag_vec(arow, k0, hi);
#pragma unroll
    for (int t = 0; t < NT; ++t) {
      v16bf b = *(const v16bf*)(Wt + (size_t)(t * 16 + lo) * Ka + k0 + (hi << 4));
      acc[t] = __builtin_amdgcn_wmma_f32_16x16x32_bf16(
          false, a, false, b, (short)0, acc[t], false, false);
    }
  }

  v8f accr[NT];
#pragma unroll
  for (int t = 0; t < NT; ++t) accr[t] = (v8f)(0.0f);
  if (ROOT) {
    const float* xrow = Xr + (size_t)mrow * Kr;
    for (int k0 = 0; k0 < Kr; k0 += 32) {
      v16bf a = load_a_frag_vec(xrow, k0, hi);
#pragma unroll
      for (int t = 0; t < NT; ++t) {
        v16bf b = *(const v16bf*)(Wrt + (size_t)(t * 16 + lo) * Kr + k0 + (hi << 4));
        accr[t] = __builtin_amdgcn_wmma_f32_16x16x32_bf16(
            false, a, false, b, (short)0, accr[t], false, false);
      }
    }
  }

#pragma unroll
  for (int t = 0; t < NT; ++t) {
#pragma unroll
    for (int i = 0; i < 8; ++i) {
      int m = tm + i + (hi << 3);
      int nn = t * 16 + lo;
      if (m < n) {
        float vv = acc[t][i];
        if (deg != nullptr) vv /= fmaxf(deg[m], 1.0f);
        vv += accr[t][i] + bias[nn];
        if (do_elu) vv = eluf(vv);
        out[(size_t)m * COUT + nn] = vv;
      }
    }
  }
}

// ------------------------------- pooling -----------------------------------

__global__ void pool_max_kernel(const float* __restrict__ x, const int* __restrict__ cluster,
                                float* __restrict__ xp, int nsrc, int C) {
  int i = blockIdx.x * blockDim.x + threadIdx.x;
  if (i >= nsrc * C) return;
  int node = i / C, c = i - node * C;
  float v = x[i];
  float* addr = xp + (long)cluster[node] * C + c;
  // float atomic max via ordered int/uint trick (init must be <= all values)
  if (v >= 0.0f) atomicMax((int*)addr, __float_as_int(v));
  else           atomicMin((unsigned int*)addr, __float_as_uint(v));
}

__global__ void pool_pos_kernel(const float* __restrict__ pos, const int* __restrict__ cluster,
                                float* __restrict__ psum, float* __restrict__ cnt, int nsrc) {
  int i = blockIdx.x * blockDim.x + threadIdx.x;
  if (i >= nsrc) return;
  int cl = cluster[i];
  atomicAdd(&cnt[cl], 1.0f);
  atomicAdd(&psum[cl * 3 + 0], pos[i * 3 + 0]);
  atomicAdd(&psum[cl * 3 + 1], pos[i * 3 + 1]);
  atomicAdd(&psum[cl * 3 + 2], pos[i * 3 + 2]);
}

__global__ void pos_div_kernel(float* __restrict__ psum, const float* __restrict__ cnt, int ndst) {
  int i = blockIdx.x * blockDim.x + threadIdx.x;
  if (i >= ndst) return;
  float c = cnt[i];
  float inv = c > 0.0f ? 1.0f / c : 0.0f;
  psum[i * 3 + 0] *= inv;
  psum[i * 3 + 1] *= inv;
  psum[i * 3 + 2] *= inv;
}

__global__ void gather_kernel(const float* __restrict__ src, const int* __restrict__ idx,
                              float* __restrict__ dst, int n, int C) {
  int i = blockIdx.x * blockDim.x + threadIdx.x;
  if (i >= n * C) return;
  int node = i / C, c = i - node * C;
  dst[i] = src[(long)idx[node] * C + c];
}

// FC 64->2 + ELU + log_softmax
__global__ void head_kernel(const float* __restrict__ u0, const float* __restrict__ fw2,
                            const float* __restrict__ fb2, float* __restrict__ out, int n) {
  int i = blockIdx.x * blockDim.x + threadIdx.x;
  if (i >= n) return;
  float a = fb2[0], b = fb2[1];
#pragma unroll 8
  for (int c = 0; c < 64; ++c) {
    float v = u0[(long)i * 64 + c];
    a += v * fw2[c * 2 + 0];
    b += v * fw2[c * 2 + 1];
  }
  a = eluf(a); b = eluf(b);
  float m = fmaxf(a, b);
  float lse = m + logf(expf(a - m) + expf(b - m));
  out[i * 2 + 0] = a - lse;
  out[i * 2 + 1] = b - lse;
}

// --------------------------------- host ------------------------------------

extern "C" void kernel_launch(void* const* d_in, const int* in_sizes, int n_in,
                              void* d_out, int out_size, void* d_ws, size_t ws_size,
                              hipStream_t stream) {
  (void)n_in; (void)out_size;
  const float* x   = (const float*)d_in[0];
  const float* pos = (const float*)d_in[1];
  const float* Wf[8]; const float* Rf[8]; const float* Bf[8];
  for (int i = 1; i <= 7; ++i) {
    Wf[i] = (const float*)d_in[2 + (i - 1) * 3];
    Rf[i] = (const float*)d_in[3 + (i - 1) * 3];
    Bf[i] = (const float*)d_in[4 + (i - 1) * 3];
  }
  const float* fw1 = (const float*)d_in[23];
  const float* fb1 = (const float*)d_in[24];
  const float* fw2 = (const float*)d_in[25];
  const float* fb2 = (const float*)d_in[26];
  const int* e[4]; int E[4];
  for (int l = 0; l < 4; ++l) { e[l] = (const int*)d_in[27 + l]; E[l] = in_sizes[27 + l] / 2; }
  const int* cl1 = (const int*)d_in[31];
  const int* cl2 = (const int*)d_in[32];
  const int* cl3 = (const int*)d_in[33];
  const int N0 = in_sizes[0];
  const int n1 = in_sizes[32];   // len(cluster2)
  const int n2 = in_sizes[33];   // len(cluster3)
  const int n3b = n2;            // upper bound on n3 (device scalar unreadable in capture);
                                 // padded rows are initialized benignly and never consumed.

  const int ci[8] = {0, 1, 32, 64, 64, 64, 64, 64};
  const int co[8] = {0, 32, 64, 64, 64, 64, 64, 64};
  int kp[8], rp[8];                               // padded Ka / Kr per conv
  for (int i = 1; i <= 7; ++i) {
    kp[i] = ((K3Q * ci[i]) + 31) & ~31;           // 128, 4000, 8000...
    rp[i] = (ci[i] + 31) & ~31;                   // 32, 32, 64...
  }

  // bump allocator over workspace
  size_t off = 0;
  auto alloc = [&](size_t bytes) -> void* {
    off = (off + 255) & ~(size_t)255;
    void* p = (char*)d_ws + off;
    off += bytes;
    return p;
  };
  auto allocF = [&](size_t n) { return (float*)alloc(n * sizeof(float)); };
  auto allocB = [&](size_t n) { return (bf16_t*)alloc(n * sizeof(bf16_t)); };

  float* ps0  = allocF((size_t)E[0] * 3);
  float* ps1  = allocF((size_t)E[1] * 3);
  float* ps2  = allocF((size_t)E[2] * 3);
  float* ps3  = allocF((size_t)E[3] * 3);
  float* pos1 = allocF((size_t)n1 * 3);
  float* pos2 = allocF((size_t)n2 * 3);
  float* pos3 = allocF((size_t)n3b * 3);
  float* cnt  = allocF((size_t)(n1 > n2 ? n1 : n2));
  float* degb = allocF((size_t)N0);
  float* xpad = allocF((size_t)N0 * 32);          // conv1 root input, padded Kr=32
  float* hA   = allocF((size_t)N0 * 32);
  float* h1p  = allocF((size_t)n1 * 32);
  float* h1   = allocF((size_t)n1 * 64);
  float* h2p  = allocF((size_t)n2 * 64);
  float* h2   = allocF((size_t)n2 * 64);
  float* h3p  = allocF((size_t)n3b * 64);
  float* h3   = allocF((size_t)n3b * 64);
  float* h3f  = allocF((size_t)n3b * 64);
  float* g2   = allocF((size_t)n2 * 64);
  float* u2   = allocF((size_t)n2 * 64);
  float* g1   = allocF((size_t)n1 * 64);
  float* u1   = allocF((size_t)n1 * 64);
  float* g0   = allocF((size_t)N0 * 64);
  float* u0   = allocF((size_t)N0 * 64);

  bf16_t* Wb[8]; bf16_t* Rb[8];     // transposed+padded: co x kp, co x rp
  for (int i = 1; i <= 7; ++i) {
    Wb[i] = allocB((size_t)co[i] * kp[i]);
    Rb[i] = allocB((size_t)co[i] * rp[i]);
  }
  bf16_t* fw1b = allocB(64 * 64);   // transposed: 64 x 64

  off = (off + 255) & ~(size_t)255;
  size_t abytes = (ws_size > off) ? (ws_size - off) : 0;
  float* Abuf = (float*)((char*)d_ws + off);

  const int TB = 256;
  auto g1d = [&](long nthr) { return dim3((unsigned)((nthr + TB - 1) / TB)); };

  // transpose+convert weights to bf16 (cout x Kpad); zero the pad first
  for (int i = 1; i <= 7; ++i) {
    hipMemsetAsync(Wb[i], 0, (size_t)co[i] * kp[i] * sizeof(bf16_t), stream);
    hipMemsetAsync(Rb[i], 0, (size_t)co[i] * rp[i] * sizeof(bf16_t), stream);
    long nw = (long)K3Q * ci[i] * co[i];
    f32_to_bf16_T_kernel<<<g1d(nw), TB, 0, stream>>>(Wf[i], Wb[i], K3Q * ci[i], co[i], kp[i]);
    long nr = (long)ci[i] * co[i];
    f32_to_bf16_T_kernel<<<g1d(nr), TB, 0, stream>>>(Rf[i], Rb[i], ci[i], co[i], rp[i]);
  }
  f32_to_bf16_T_kernel<<<g1d(64 * 64), TB, 0, stream>>>(fw1, fw1b, 64, 64, 64);

  // conv1 root input: x padded to (N0 x 32)
  hipMemsetAsync(xpad, 0, (size_t)N0 * 32 * sizeof(float), stream);
  pad_x_kernel<<<g1d(N0), TB, 0, stream>>>(x, xpad, N0);

  // spline conv: deg -> (chunked) zero A, scatter, WMMA GEMM + root + bias + ELU
  // xin: n x cin (conv input, also root input after padding); xr: n x rp
  auto run_conv = [&](const float* xin, const float* xr, int n, int cin, int idx,
                      const int* edg, int En, const float* psv, float* outb) {
    const int* erow = edg;
    const int* ecol = edg + En;
    const int cout = co[idx], Ka = kp[idx], Kr = rp[idx];
    hipMemsetAsync(degb, 0, (size_t)n * sizeof(float), stream);
    deg_count_kernel<<<g1d(En), TB, 0, stream>>>(erow, degb, En);
    size_t rowbytes = (size_t)Ka * sizeof(float);
    long cap = (long)(abytes / rowbytes);
    cap &= ~15L;
    if (cap < 16) cap = 16;
    for (long start = 0; start < n; start += cap) {
      int cn = (int)(((long)n - start < cap) ? ((long)n - start) : cap);
      hipMemsetAsync(Abuf, 0, (size_t)cn * rowbytes, stream);
      if (cin == 1) {
        spline_scatter_c1_kernel<<<g1d(En), TB, 0, stream>>>(
            xin, erow, ecol, psv, Abuf, (int)start, cn, En, Ka);
      } else {
        spline_scatter_kernel<<<dim3((unsigned)En), dim3((unsigned)cin), 0, stream>>>(
            xin, erow, ecol, psv, Abuf, (int)start, cn, En, cin, Ka);
      }
      dim3 gg((unsigned)((cn + 15) / 16));
      if (cout == 32) {
        wmma_gemm_vec_kernel<2, true><<<gg, 32, 0, stream>>>(
            Abuf, Wb[idx], degb + start, xr + start * Kr, Rb[idx], Bf[idx],
            outb + start * cout, cn, Ka, Kr, 1);
      } else {
        wmma_gemm_vec_kernel<4, true><<<gg, 32, 0, stream>>>(
            Abuf, Wb[idx], degb + start, xr + start * Kr, Rb[idx], Bf[idx],
            outb + start * cout, cn, Ka, Kr, 1);
      }
    }
  };

  auto run_pool = [&](const float* xin, const float* pin, const int* clus,
                      int nsrc, int ndst, int C, float* xout, float* pout) {
    // ELU outputs are > -1, every real cluster is non-empty -> -1 is a safe identity
    fill_kernel<<<g1d((long)ndst * C), TB, 0, stream>>>(xout, -1.0f, ndst * C);
    pool_max_kernel<<<g1d((long)nsrc * C), TB, 0, stream>>>(xin, clus, xout, nsrc, C);
    hipMemsetAsync(pout, 0, (size_t)ndst * 3 * sizeof(float), stream);
    hipMemsetAsync(cnt, 0, (size_t)ndst * sizeof(float), stream);
    pool_pos_kernel<<<g1d(nsrc), TB, 0, stream>>>(pin, clus, pout, cnt, nsrc);
    pos_div_kernel<<<g1d(ndst), TB, 0, stream>>>(pout, cnt, ndst);
  };

  // ----------------------------- forward pass -----------------------------
  pseudo_kernel<<<g1d(E[0]), TB, 0, stream>>>(pos, e[0], e[0] + E[0], ps0, E[0]);
  run_conv(x, xpad, N0, 1, 1, e[0], E[0], ps0, hA);
  run_pool(hA, pos, cl1, N0, n1, 32, h1p, pos1);

  pseudo_kernel<<<g1d(E[1]), TB, 0, stream>>>(pos1, e[1], e[1] + E[1], ps1, E[1]);
  run_conv(h1p, h1p, n1, 32, 2, e[1], E[1], ps1, h1);
  run_pool(h1, pos1, cl2, n1, n2, 64, h2p, pos2);

  pseudo_kernel<<<g1d(E[2]), TB, 0, stream>>>(pos2, e[2], e[2] + E[2], ps2, E[2]);
  run_conv(h2p, h2p, n2, 64, 3, e[2], E[2], ps2, h2);
  run_pool(h2, pos2, cl3, n2, n3b, 64, h3p, pos3);

  pseudo_kernel<<<g1d(E[3]), TB, 0, stream>>>(pos3, e[3], e[3] + E[3], ps3, E[3]);
  run_conv(h3p, h3p, n3b, 64, 4, e[3], E[3], ps3, h3);

  // bottleneck FC (WMMA vec path: Ka=64, no deg / no root)
  wmma_gemm_vec_kernel<4, false><<<dim3((unsigned)((n3b + 15) / 16)), 32, 0, stream>>>(
      h3, fw1b, nullptr, nullptr, nullptr, fb1, h3f, n3b, 64, 0, 1);

  gather_kernel<<<g1d((long)n2 * 64), TB, 0, stream>>>(h3f, cl3, g2, n2, 64);
  run_conv(g2, g2, n2, 64, 5, e[2], E[2], ps2, u2);

  gather_kernel<<<g1d((long)n1 * 64), TB, 0, stream>>>(u2, cl2, g1, n1, 64);
  run_conv(g1, g1, n1, 64, 6, e[1], E[1], ps1, u1);

  gather_kernel<<<g1d((long)N0 * 64), TB, 0, stream>>>(u1, cl1, g0, N0, 64);
  run_conv(g0, g0, N0, 64, 7, e[0], E[0], ps0, u0);

  head_kernel<<<g1d(N0), TB, 0, stream>>>(u0, fw2, fb2, (float*)d_out, N0);
}
